// HybridViT_32830730011484
// MI455X (gfx1250) — matrix operand order repaired and checked by reading, not compiled, not run on hardware
//
#include <hip/hip_runtime.h>
#include <math.h>

typedef __bf16 bf16;
typedef __attribute__((ext_vector_type(16))) __bf16 v16bf;
typedef __attribute__((ext_vector_type(8)))  float  v8f;

// ---------------------------------------------------------------------------
// Generic GEMM:  C[M,N] = act(A[M,K] @ B[K,N] + bias) + resid
// N is a COMPILE-TIME template parameter so that per-element B offsets
// (e*N*4 bytes) become instruction immediates:
//   global_load_b32 v, voffs[s], s[base] offset:e*N*4
// -> no per-load address ALU, no xcnt chains, clause-able loads.
// Requirements: M % 16 == 0, K % 32 == 0 (all call sites). N ragged OK.
// One wave -> 16x64 of C (4 subtiles), one A fragment per K-step.
// act: 0 = none, 1 = exact GELU
// ---------------------------------------------------------------------------
template <int N>
__launch_bounds__(256)
__global__ void gemm_bf16_wmma(const float* __restrict__ A, const float* __restrict__ Bm,
                               const float* __restrict__ bias, const float* __restrict__ resid,
                               float* __restrict__ C, int M, int K, int act) {
    const int lane = threadIdx.x & 31;
    const int wave = threadIdx.x >> 5;
    const int m0   = blockIdx.x * 16;
    const int n0   = (blockIdx.y * 8 + wave) * 64;   // 4 x 16 columns per wave
    if (n0 >= N) return;                             // uniform per wave
    const int half = lane >> 4;
    const int nl   = lane & 15;
    const int m    = m0 + nl;

    int  ncol[4]; bool nok[4];
    #pragma unroll
    for (int s = 0; s < 4; ++s) {
        int n   = n0 + s * 16 + nl;
        nok[s]  = (n < N);
        ncol[s] = nok[s] ? n : (N - 1);              // clamped, always in-range
    }
    int voffs[4];                                    // per-lane, k-invariant
    #pragma unroll
    for (int s = 0; s < 4; ++s) voffs[s] = half * 16 * N + ncol[s];

    const float* ap  = A + (size_t)m * K;            // per-lane A row pointer
    const int    aoff = half * 16;

    v8f acc[4] = {};
    for (int k0 = 0; k0 < K; k0 += 32) {
        if (k0 + 32 < K) __builtin_prefetch(&ap[k0 + 32], 0, 1);
        v16bf af;
        #pragma unroll
        for (int e = 0; e < 16; e += 2) {
            int k = k0 + aoff + e;
            af[e]   = (bf16)ap[k];
            af[e+1] = (bf16)ap[k + 1];
        }
        const float* bp = Bm + (size_t)k0 * N;       // uniform SGPR base
        #pragma unroll
        for (int s = 0; s < 4; ++s) {
            float xv[16];
            #pragma unroll
            for (int e = 0; e < 16; ++e)
                xv[e] = bp[voffs[s] + e * N];        // saddr + voffset + CONST imm
            v16bf bfm;
            #pragma unroll
            for (int e = 0; e < 16; ++e) {
                float x = nok[s] ? xv[e] : 0.f;
                bfm[e] = (bf16)x;
            }
            acc[s] = __builtin_amdgcn_wmma_f32_16x16x32_bf16(false, af, false, bfm,
                                                             (short)0, acc[s], false, false);
        }
    }

    #pragma unroll
    for (int s = 0; s < 4; ++s) {
        int n = n0 + s * 16 + nl;
        if (n < N) {
            #pragma unroll
            for (int r = 0; r < 8; ++r) {
                int mr = m0 + half * 8 + r;
                float t = acc[s][r];
                if (bias)     t += bias[n];
                if (act == 1) t = 0.5f * t * (1.0f + erff(t * 0.70710678118654752440f));
                if (resid)    t += resid[(size_t)mr * N + n];
                C[(size_t)mr * N + n] = t;
            }
        }
    }
}

// ---------------------------------------------------------------------------
// Implicit-GEMM 7x7 stride-2 VALID conv + ReLU, per patch (blockIdx.z).
// M = Cout, N = Hout*Wout, K = CIN*49. Weights OIHW row-major == A[M,K].
// im2col offsets delta[k] in LDS (ds_load_b128 when K%32==0);
// B(k,n) = IN[sbase(n) + delta(k)] via independent v_add + saddr loads.
// One wave -> 16x64 of the output (4 subtiles).
// ---------------------------------------------------------------------------
template <int CIN>
__launch_bounds__(256)
__global__ void conv7s2_wmma(const float* __restrict__ IN, const float* __restrict__ W,
                             float* __restrict__ OUT, int Hin, int Win, int Cout,
                             int Hout, int Wout) {
    constexpr int K = CIN * 49;
    __shared__ __align__(16) int delta[(K + 15) & ~15];
    for (int k = threadIdx.x; k < K; k += 256) {
        int ci = k / 49; int r = k - ci * 49;
        int kh = r / 7;  int kw = r - kh * 7;
        delta[k] = ci * Hin * Win + kh * Win + kw;
    }
    __syncthreads();

    const int patch = blockIdx.z;
    const int lane  = threadIdx.x & 31;
    const int wave  = threadIdx.x >> 5;
    const int m0    = blockIdx.x * 16;
    const int Npix  = Hout * Wout;
    const int n0    = (blockIdx.y * 8 + wave) * 64;
    if (n0 >= Npix) return;                          // after the barrier
    const int half = lane >> 4;
    const int nl   = lane & 15;
    const int m    = m0 + nl;

    int  sbase[4]; bool nok[4];
    #pragma unroll
    for (int s = 0; s < 4; ++s) {
        int n  = n0 + s * 16 + nl;
        nok[s] = (n < Npix);
        int nc = nok[s] ? n : (Npix - 1);
        int oh = nc / Wout; int ow = nc - oh * Wout;
        sbase[s] = (oh * 2) * Win + ow * 2;
    }
    const float* inP = IN + (size_t)patch * CIN * Hin * Win;   // uniform base
    const float* wp  = W + (size_t)m * K;                      // per-lane A row
    const int    aoff = half * 16;

    v8f acc[4] = {};
    for (int k0 = 0; k0 < K; k0 += 32) {
        v16bf af;
        #pragma unroll
        for (int e = 0; e < 16; e += 2) {
            int k = k0 + aoff + e;
            if constexpr (K % 32 == 0) {
                af[e]   = (bf16)wp[k];
                af[e+1] = (bf16)wp[k + 1];
            } else {
                float x0 = wp[k     < K ? k     : K - 1];
                float x1 = wp[k + 1 < K ? k + 1 : K - 1];
                af[e]   = (bf16)(k     < K ? x0 : 0.f);
                af[e+1] = (bf16)(k + 1 < K ? x1 : 0.f);
            }
        }
        const int kb = k0 + half * 16;
        int dd[16]; bool kok[16];
        if constexpr (K % 32 == 0) {
            const int4* dp = (const int4*)(delta + kb);        // 16B-aligned (kb % 16 == 0)
            #pragma unroll
            for (int i = 0; i < 4; ++i) {                      // 4 x ds_load_b128
                int4 t = dp[i];
                dd[4*i] = t.x; dd[4*i+1] = t.y; dd[4*i+2] = t.z; dd[4*i+3] = t.w;
                kok[4*i] = kok[4*i+1] = kok[4*i+2] = kok[4*i+3] = true;
            }
        } else {
            #pragma unroll
            for (int e = 0; e < 16; ++e) {
                int k  = kb + e;
                kok[e] = (k < K);
                dd[e]  = delta[kok[e] ? k : (K - 1)];
            }
        }
        #pragma unroll
        for (int s = 0; s < 4; ++s) {
            float xv[16];
            #pragma unroll
            for (int e = 0; e < 16; ++e)
                xv[e] = inP[sbase[s] + dd[e]];                 // independent v_add + saddr load
            v16bf bfm;
            #pragma unroll
            for (int e = 0; e < 16; ++e) {
                float x = (kok[e] && nok[s]) ? xv[e] : 0.f;
                bfm[e] = (bf16)x;
            }
            acc[s] = __builtin_amdgcn_wmma_f32_16x16x32_bf16(false, af, false, bfm,
                                                             (short)0, acc[s], false, false);
        }
    }

    float* outP = OUT + (size_t)patch * Cout * Npix;
    #pragma unroll
    for (int s = 0; s < 4; ++s) {
        int n = n0 + s * 16 + nl;
        if (n < Npix) {
            #pragma unroll
            for (int r = 0; r < 8; ++r) {
                int mr = m0 + half * 8 + r;
                float t = acc[s][r];
                outP[(size_t)mr * Npix + n] = t > 0.f ? t : 0.f;   // fused ReLU
            }
        }
    }
}

// ---------------------------------------------------------------------------
// MaxPool2d(kernel=7, stride=1, padding=1): Hout = Hin - 4
// ---------------------------------------------------------------------------
__global__ void maxpool7s1p1(const float* __restrict__ IN, float* __restrict__ OUT,
                             int NC, int Hin, int Win, int Hout, int Wout) {
    size_t idx   = (size_t)blockIdx.x * blockDim.x + threadIdx.x;
    size_t total = (size_t)NC * Hout * Wout;
    if (idx >= total) return;
    int ow = (int)(idx % Wout); size_t t = idx / Wout;
    int oh = (int)(t % Hout);   int c  = (int)(t / Hout);
    const float* p = IN + (size_t)c * Hin * Win;
    float mx = -INFINITY;
    for (int dy = -1; dy <= 5; ++dy) {
        int y = oh + dy; if (y < 0 || y >= Hin) continue;
        for (int dx = -1; dx <= 5; ++dx) {
            int x = ow + dx; if (x < 0 || x >= Win) continue;
            float v = p[(size_t)y * Win + x];
            mx = v > mx ? v : mx;
        }
    }
    OUT[idx] = mx;
}

// img [16,1,880,880] -> patches [256,1,220,220]; patch = b*16 + gh*4 + gw
__global__ void patchify_k(const float* __restrict__ img, float* __restrict__ out) {
    size_t idx = (size_t)blockIdx.x * blockDim.x + threadIdx.x;
    if (idx >= 256ull * 220 * 220) return;
    int c = (int)(idx % 220); size_t t = idx / 220;
    int r = (int)(t % 220);   int p  = (int)(t / 220);
    int b = p >> 4; int gh = (p >> 2) & 3; int gw = p & 3;
    out[idx] = img[((size_t)b * 880 + gh * 220 + r) * 880 + gw * 220 + c];
}

// x[b][0] = cls + pos[0]; x[b][1+t] = flat[b*16+t] + pos[1+t]
__global__ void assemble_tokens(const float* __restrict__ flat, const float* __restrict__ cls,
                                const float* __restrict__ pos, float* __restrict__ X) {
    int row = blockIdx.x; int b = row / 17; int t = row % 17;
    for (int i = threadIdx.x; i < 512; i += 256) {
        float v = pos[t * 512 + i];
        v += (t == 0) ? cls[i] : flat[((size_t)b * 16 + (t - 1)) * 512 + i];
        X[(size_t)row * 512 + i] = v;
    }
}

// LayerNorm over 512; one block per row; input rows strided by in_stride elements
__global__ void layernorm_k(const float* __restrict__ X, float* __restrict__ Y,
                            const float* __restrict__ g, const float* __restrict__ b,
                            int in_stride) {
    int row = blockIdx.x;
    const float* x = X + (size_t)row * in_stride;
    float* y = Y + (size_t)row * 512;
    __shared__ float ssum[256], ssq[256];
    int t = threadIdx.x;
    float a0 = x[t], a1 = x[t + 256];
    ssum[t] = a0 + a1; ssq[t] = a0 * a0 + a1 * a1;
    __syncthreads();
    for (int s = 128; s > 0; s >>= 1) {
        if (t < s) { ssum[t] += ssum[t + s]; ssq[t] += ssq[t + s]; }
        __syncthreads();
    }
    float mean = ssum[0] * (1.f / 512.f);
    float var  = ssq[0] * (1.f / 512.f) - mean * mean;
    float rstd = rsqrtf(var + 1e-5f);
    y[t]       = (a0 - mean) * rstd * g[t]       + b[t];
    y[t + 256] = (a1 - mean) * rstd * g[t + 256] + b[t + 256];
}

// Attention over 17 tokens. One wave per (b, head); thread q < 17 owns query q.
// QKV row layout per token: [q(512) | k(512) | v(512)], head h at offset h*64.
__global__ void attention_k(const float* __restrict__ QKV, float* __restrict__ O) {
    int bh = blockIdx.x; int b = bh >> 3; int h = bh & 7;
    int q  = threadIdx.x;
    const float* base = QKV + (size_t)b * 17 * 1536;
    if (q < 17) {
        const float* qp = base + (size_t)q * 1536 + h * 64;
        float s[17];
        float mx = -INFINITY;
        for (int j = 0; j < 17; ++j) {
            const float* kp = base + (size_t)j * 1536 + 512 + h * 64;
            float d = 0.f;
            for (int e = 0; e < 64; ++e) d += qp[e] * kp[e];
            d *= 0.125f;                       // DH^-0.5
            s[j] = d; mx = d > mx ? d : mx;
        }
        float sum = 0.f;
        for (int j = 0; j < 17; ++j) { s[j] = __expf(s[j] - mx); sum += s[j]; }
        float inv = 1.f / sum;
        float* op = O + ((size_t)b * 17 + q) * 512 + h * 64;
        for (int e = 0; e < 64; ++e) {
            float acc = 0.f;
            for (int j = 0; j < 17; ++j)
                acc += s[j] * base[(size_t)j * 1536 + 1024 + h * 64 + e];
            op[e] = acc * inv;
        }
    }
}

static inline dim3 gemm_grid(int M, int N) {
    int mt  = (M + 15) / 16;
    int nt  = (N + 63) / 64;                 // 64 columns per wave
    return dim3((unsigned)mt, (unsigned)((nt + 7) / 8), 1);
}

extern "C" void kernel_launch(void* const* d_in, const int* in_sizes, int n_in,
                              void* d_out, int out_size, void* d_ws, size_t ws_size,
                              hipStream_t stream) {
    (void)in_sizes; (void)n_in; (void)out_size; (void)ws_size;
    const float* img    = (const float*)d_in[0];
    const float* c1w    = (const float*)d_in[1];
    const float* c2w    = (const float*)d_in[2];
    const float* c3w    = (const float*)d_in[3];
    const float* flat_w = (const float*)d_in[4];
    const float* flat_b = (const float*)d_in[5];
    const float* cls    = (const float*)d_in[6];
    const float* pos    = (const float*)d_in[7];
    const float* ln1g   = (const float*)d_in[8];
    const float* ln1b   = (const float*)d_in[9];
    const float* qkvw   = (const float*)d_in[10];
    const float* outw   = (const float*)d_in[11];
    const float* outb   = (const float*)d_in[12];
    const float* ln2g   = (const float*)d_in[13];
    const float* ln2b   = (const float*)d_in[14];
    const float* ff1w   = (const float*)d_in[15];
    const float* ff1b   = (const float*)d_in[16];
    const float* ff2w   = (const float*)d_in[17];
    const float* ff2b   = (const float*)d_in[18];
    const float* hlng   = (const float*)d_in[19];
    const float* hlnb   = (const float*)d_in[20];
    const float* headw  = (const float*)d_in[21];
    const float* headb  = (const float*)d_in[22];

    float* wsf  = (float*)d_ws;
    float* bufA = wsf;                         // up to 93,790,208 floats (conv outs)
    float* bufB = wsf + 94000000ull;           // up to 86,908,928 floats (patches/pools)
    float* S    = wsf + 181000000ull;          // small transformer buffers
    float* flatout = S;                        // 256*512
    float* xbuf = S + 131072;                  // 272*512
    float* xalt = xbuf + 139264;               // 272*512
    float* hbuf = xalt + 139264;               // 272*512
    float* qkv  = hbuf + 139264;               // 272*1536
    float* abuf = qkv  + 417792;               // 272*512
    float* fbuf = abuf + 139264;               // 272*2048
    float* clsb = fbuf + 557056;               // 16*512

    // ---- CNN stem ----
    {
        size_t total = 256ull * 220 * 220;
        patchify_k<<<dim3((unsigned)((total + 255) / 256)), 256, 0, stream>>>(img, bufB);
    }
    // conv1: 1->32, 220 -> 107 ; N=11449, tiles64 = 179
    conv7s2_wmma<1><<<dim3(2, (179 + 7) / 8, 256), 256, 0, stream>>>(
        bufB, c1w, bufA, 220, 220, 32, 107, 107);
    {
        size_t total = 256ull * 32 * 103 * 103;
        maxpool7s1p1<<<dim3((unsigned)((total + 255) / 256)), 256, 0, stream>>>(
            bufA, bufB, 256 * 32, 107, 107, 103, 103);
    }
    // conv2: 32->64, 103 -> 49 ; N=2401, tiles64 = 38
    conv7s2_wmma<32><<<dim3(4, (38 + 7) / 8, 256), 256, 0, stream>>>(
        bufB, c2w, bufA, 103, 103, 64, 49, 49);
    {
        size_t total = 256ull * 64 * 45 * 45;
        maxpool7s1p1<<<dim3((unsigned)((total + 255) / 256)), 256, 0, stream>>>(
            bufA, bufB, 256 * 64, 49, 49, 45, 45);
    }
    // conv3: 64->64, 45 -> 20 ; N=400, tiles64 = 7
    conv7s2_wmma<64><<<dim3(4, 1, 256), 256, 0, stream>>>(
        bufB, c3w, bufA, 45, 45, 64, 20, 20);
    {
        size_t total = 256ull * 64 * 16 * 16;
        maxpool7s1p1<<<dim3((unsigned)((total + 255) / 256)), 256, 0, stream>>>(
            bufA, bufB, 256 * 64, 20, 20, 16, 16);
    }

    // flatten: [256,16384] @ [16384,512] + flat_b
    gemm_bf16_wmma<512><<<gemm_grid(256, 512), 256, 0, stream>>>(
        bufB, flat_w, flat_b, nullptr, flatout, 256, 16384, 0);

    assemble_tokens<<<272, 256, 0, stream>>>(flatout, cls, pos, xbuf);

    // ---- transformer (6 layers) ----
    for (int l = 0; l < 6; ++l) {
        layernorm_k<<<272, 256, 0, stream>>>(xbuf, hbuf, ln1g + l * 512, ln1b + l * 512, 512);
        gemm_bf16_wmma<1536><<<gemm_grid(272, 1536), 256, 0, stream>>>(
            hbuf, qkvw + (size_t)l * 512 * 1536, nullptr, nullptr, qkv, 272, 512, 0);
        attention_k<<<128, 32, 0, stream>>>(qkv, abuf);
        gemm_bf16_wmma<512><<<gemm_grid(272, 512), 256, 0, stream>>>(
            abuf, outw + (size_t)l * 512 * 512, outb + l * 512, xbuf, xalt, 272, 512, 0);
        layernorm_k<<<272, 256, 0, stream>>>(xalt, hbuf, ln2g + l * 512, ln2b + l * 512, 512);
        gemm_bf16_wmma<2048><<<gemm_grid(272, 2048), 256, 0, stream>>>(
            hbuf, ff1w + (size_t)l * 512 * 2048, ff1b + l * 2048, nullptr, fbuf, 272, 512, 1);
        gemm_bf16_wmma<512><<<gemm_grid(272, 512), 256, 0, stream>>>(
            fbuf, ff2w + (size_t)l * 2048 * 512, ff2b + l * 512, xalt, xbuf, 272, 2048, 0);
    }

    // ---- head ----
    layernorm_k<<<16, 256, 0, stream>>>(xbuf, clsb, hlng, hlnb, 17 * 512);
    gemm_bf16_wmma<1000><<<gemm_grid(16, 1000), 256, 0, stream>>>(
        clsb, headw, headb, nullptr, (float*)d_out, 16, 512, 0);
}
